// ONOBlock_linear_attn_69999376990759
// MI455X (gfx1250) — compile-verified, matrix operand order checked by reading, not run on hardware
//
#include <hip/hip_runtime.h>

// ---------------------------------------------------------------------------
// ONO block with linear attention, fused for gfx1250 (MI455X).
// B=4, N=8192, C=128, H=8, d=16.  All GEMMs via v_wmma_f32_16x16x32_f16.
// ---------------------------------------------------------------------------

typedef __attribute__((ext_vector_type(16))) _Float16 v16h;
typedef __attribute__((ext_vector_type(8)))  _Float16 v8h;
typedef __attribute__((ext_vector_type(8)))  float    v8f;

#define TPB 128            // 4 waves per workgroup
#define ROWS_PER_WG 64     // 16 tokens per wave
#define NTOK 32768         // B*N
#define NPERB 8192         // N
#define NBLK (NTOK / ROWS_PER_WG)   // 512

// workspace layout (float offsets)
#define OFF_XLN1  0ull
#define OFF_Q1S   4194304ull
#define OFF_FXN   8388608ull
#define OFF_Q2S   12582912ull
#define OFF_CTX1  16777216ull
#define OFF_KSUM1 16785408ull
#define OFF_CTX2  16785920ull
#define OFF_KSUM2 16794112ull
#define OFF_WFRAG_BYTES 67178496ull   // (OFF_KSUM2+512)*4, 32B aligned

struct WPtrs { const float* w[10]; };

// ------------------------------ helpers ------------------------------------

__device__ __forceinline__ float wave_sum32(float v) {
#pragma unroll
  for (int o = 16; o; o >>= 1) v += __shfl_xor(v, o, 32);
  return v;
}

__device__ __forceinline__ float gelu_exact(float x) {
  return 0.5f * x * (1.0f + erff(x * 0.70710678118654752f));
}

// LayerNorm 16 rows of 128; src rows contiguous (global or LDS).
// Writes f16 into ah tile, optionally f32 rows to dstF.
__device__ __forceinline__ void ln16(const float* src, float* dstF,
                                     _Float16* ah, const float* gw,
                                     const float* gb, int lane) {
  float4 w4 = *(const float4*)(gw + lane * 4);
  float4 b4 = *(const float4*)(gb + lane * 4);
#pragma unroll 1
  for (int m = 0; m < 16; ++m) {
    float4 x = *(const float4*)(src + m * 128 + lane * 4);
    float mu = wave_sum32(x.x + x.y + x.z + x.w) * (1.0f / 128.0f);
    float d0 = x.x - mu, d1 = x.y - mu, d2 = x.z - mu, d3 = x.w - mu;
    float var = wave_sum32(d0 * d0 + d1 * d1 + d2 * d2 + d3 * d3) * (1.0f / 128.0f);
    float rs = rsqrtf(var + 1e-6f);
    float o0 = d0 * rs * w4.x + b4.x, o1 = d1 * rs * w4.y + b4.y;
    float o2 = d2 * rs * w4.z + b4.z, o3 = d3 * rs * w4.w + b4.w;
    if (dstF) {
      float4 o = {o0, o1, o2, o3};
      *(float4*)(dstF + m * 128 + lane * 4) = o;
    }
    _Float16* ap = ah + m * 128 + lane * 4;
    ap[0] = (_Float16)o0; ap[1] = (_Float16)o1;
    ap[2] = (_Float16)o2; ap[3] = (_Float16)o3;
  }
}

// A fragment for v_wmma_f32_16x16x32_f16 from a row-major 16x128 f16 LDS tile.
// ISA layout: lanes 0-15 -> K base 0, lanes 16-31 -> K base 8;
// halves 0..7 -> K=base+i, halves 8..15 -> K=16+base+(i-8).
__device__ __forceinline__ v16h load_afrag(const _Float16* ah, int lane, int kc) {
  int m = lane & 15, base = (lane >> 4) * 8;
  const _Float16* p = ah + m * 128 + kc * 32 + base;
  v8h lo = *(const v8h*)p;
  v8h hi = *(const v8h*)(p + 16);
  v16h a;
#pragma unroll
  for (int i = 0; i < 8; ++i) { a[i] = lo[i]; a[i + 8] = hi[i]; }
  return a;
}

// D = A(16x128) * W(128x128) + bias.  W pre-packed as B fragments (f16).
// acc[ct][r]: ct = 16-col tile (== head for QKV), r = D-fragment vgpr.
__device__ __forceinline__ void gemm_tile(const _Float16* ah, const _Float16* wf,
                                          const float* bias, float acc[8][8],
                                          int lane) {
  v16h afr[4];
#pragma unroll
  for (int kc = 0; kc < 4; ++kc) afr[kc] = load_afrag(ah, lane, kc);
#pragma unroll
  for (int ct = 0; ct < 8; ++ct) {
    v8f c = {0.f, 0.f, 0.f, 0.f, 0.f, 0.f, 0.f, 0.f};
#pragma unroll
    for (int kc = 0; kc < 4; ++kc) {
      v16h b = *(const v16h*)(wf + (size_t)((ct * 4 + kc) * 32 + lane) * 16);
      c = __builtin_amdgcn_wmma_f32_16x16x32_f16(false, afr[kc], false, b,
                                                 (short)0, c, false, false);
    }
    float bb = bias[ct * 16 + (lane & 15)];
#pragma unroll
    for (int r = 0; r < 8; ++r) acc[ct][r] = c[r] + bb;
  }
}

// softmax over n (16 columns of one col-tile == one head), per row.
// D layout puts n in 16-lane groups -> shfl width 16.
__device__ __forceinline__ void softmax_ct(float* a, int lane) {
  (void)lane;
#pragma unroll
  for (int r = 0; r < 8; ++r) {
    float x = a[r], mx = x;
#pragma unroll
    for (int o = 8; o; o >>= 1) mx = fmaxf(mx, __shfl_xor(mx, o, 16));
    float e = __expf(x - mx), s = e;
#pragma unroll
    for (int o = 8; o; o >>= 1) s += __shfl_xor(s, o, 16);
    a[r] = e / s;
  }
}

__device__ __forceinline__ void frag_to_f32(const float acc[8][8], float* tile, int lane) {
  int hi = lane >> 4, lo = lane & 15;
#pragma unroll
  for (int ct = 0; ct < 8; ++ct)
#pragma unroll
    for (int r = 0; r < 8; ++r)
      tile[(r + 8 * hi) * 128 + ct * 16 + lo] = acc[ct][r];
}

__device__ __forceinline__ void frag_to_f16(const float acc[8][8], _Float16* tile, int lane) {
  int hi = lane >> 4, lo = lane & 15;
#pragma unroll
  for (int ct = 0; ct < 8; ++ct)
#pragma unroll
    for (int r = 0; r < 8; ++r)
      tile[(r + 8 * hi) * 128 + ct * 16 + lo] = (_Float16)acc[ct][r];
}

__device__ __forceinline__ void frag_to_global(const float acc[8][8], float* g, int lane) {
  int hi = lane >> 4, lo = lane & 15;
#pragma unroll
  for (int ct = 0; ct < 8; ++ct)
#pragma unroll
    for (int r = 0; r < 8; ++r)
      g[(size_t)(r + 8 * hi) * 128 + ct * 16 + lo] = acc[ct][r];
}

__device__ __forceinline__ void frag_add_global(float acc[8][8], const float* g, int lane) {
  int hi = lane >> 4, lo = lane & 15;
#pragma unroll
  for (int ct = 0; ct < 8; ++ct)
#pragma unroll
    for (int r = 0; r < 8; ++r)
      acc[ct][r] += g[(size_t)(r + 8 * hi) * 128 + ct * 16 + lo];
}

// Stage 16x128 f32 global rows into LDS (coalesced).
__device__ __forceinline__ void stage16(const float* g, float* fs, int lane) {
#pragma unroll 1
  for (int m = 0; m < 16; ++m)
    *(float4*)(fs + m * 128 + lane * 4) = *(const float4*)(g + m * 128 + lane * 4);
}

// y[t, h*16+dv] = dinv * sum_dk q[t,h*16+dk] * ctx[h,dk,dv]; result as f16 A tile.
__device__ __forceinline__ void attn_apply(const float* fsq, const float* ctxL,
                                           const float* ksL, _Float16* ah, int lane) {
  int t = lane & 15, hf = lane >> 4;
#pragma unroll 1
  for (int h = 0; h < 8; ++h) {
    const float* q = fsq + t * 128 + h * 16;
    float dinv = 0.f;
#pragma unroll
    for (int d = 0; d < 16; ++d) dinv += q[d] * ksL[h * 16 + d];
    dinv = 1.0f / dinv;
#pragma unroll
    for (int j = 0; j < 8; ++j) {
      int dv = hf * 8 + j;
      float s = 0.f;
#pragma unroll
      for (int dk = 0; dk < 16; ++dk) s += q[dk] * ctxL[h * 256 + dk * 16 + dv];
      ah[t * 128 + h * 16 + dv] = (_Float16)(s * dinv);
    }
  }
}

// Per-WG partial of ctx[h,dk,dv] += k[t,dk]*v[t,dv] and ksum over 64 tokens,
// then global f32 atomics (ctx is only 2048 floats per batch).
__device__ __forceinline__ void accum_ctx(const float (*kt)[2048], const float (*vt)[2048],
                                          float* gctx, float* gksum, int tid) {
  for (int e = tid; e < 2048; e += TPB) {
    int h = e >> 8, dk = (e >> 4) & 15, dv = e & 15;
    float s = 0.f;
    for (int t = 0; t < 64; ++t) {
      const float* kr = kt[t >> 4] + (t & 15) * 128 + h * 16;
      const float* vr = vt[t >> 4] + (t & 15) * 128 + h * 16;
      s += kr[dk] * vr[dv];
    }
    __hip_atomic_fetch_add(gctx + e, s, __ATOMIC_RELAXED, __HIP_MEMORY_SCOPE_AGENT);
  }
  for (int e = tid; e < 128; e += TPB) {
    int h = e >> 4, dk = e & 15;
    float s = 0.f;
    for (int t = 0; t < 64; ++t) s += kt[t >> 4][(t & 15) * 128 + h * 16 + dk];
    __hip_atomic_fetch_add(gksum + e, s, __ATOMIC_RELAXED, __HIP_MEMORY_SCOPE_AGENT);
  }
}

// ------------------------------ kernels ------------------------------------

__global__ __launch_bounds__(256) void zero_kernel(float* p, int n) {
  int i = blockIdx.x * blockDim.x + threadIdx.x;
  if (i < n) p[i] = 0.f;
}

// Pack 10 f32 [K=128][N=128] weights into f16 WMMA B-fragment layout.
__global__ __launch_bounds__(256) void wconv_kernel(WPtrs wp, _Float16* wfrag) {
  int idx = blockIdx.x * blockDim.x + threadIdx.x;
  if (idx >= 10 * 16384) return;
  int s = idx >> 14, e = idx & 16383;
  int k = e >> 7, n = e & 127;
  int ct = n >> 4, n16 = n & 15;
  int kc = k >> 5, kl = k & 31;
  int sec = kl >> 4, rem = kl & 15, hi = rem >> 3, j = rem & 7;
  int i = sec * 8 + j;
  int lane = n16 + (hi << 4);
  wfrag[(size_t)s * 16384 + (size_t)((ct * 4 + kc) * 32 + lane) * 16 + i] =
      (_Float16)wp.w[s][(size_t)k * 128 + n];
}

// K1: LN1 -> Q1/K1/V1 GEMMs -> softmax(q,k) -> ctx1/ksum1 partials.
__global__ __launch_bounds__(TPB) void k1_kernel(
    const float* x_in, const float* ln1w, const float* ln1b,
    const float* q1b, const float* k1b, const float* v1b,
    const _Float16* wfrag, float* xln1, float* q1s, float* ctx1, float* ksum1) {
  __shared__ __align__(16) _Float16 ah[4][2048];
  __shared__ __align__(16) float kt[4][2048];
  __shared__ __align__(16) float vt[4][2048];
  int wv = threadIdx.x >> 5, lane = threadIdx.x & 31, tid = threadIdx.x;
  int tok0 = blockIdx.x * ROWS_PER_WG;
  int tw = tok0 + wv * 16;
  int bi = tok0 / NPERB;

  ln16(x_in + (size_t)tw * 128, xln1 + (size_t)tw * 128, ah[wv], ln1w, ln1b, lane);
  __syncthreads();

  float acc[8][8];
  // Q1 -> softmax -> store
  gemm_tile(ah[wv], wfrag + 0 * 16384, q1b, acc, lane);
#pragma unroll
  for (int ct = 0; ct < 8; ++ct) softmax_ct(acc[ct], lane);
  frag_to_global(acc, q1s + (size_t)tw * 128, lane);
  // K1 -> softmax -> kt
  gemm_tile(ah[wv], wfrag + 1 * 16384, k1b, acc, lane);
#pragma unroll
  for (int ct = 0; ct < 8; ++ct) softmax_ct(acc[ct], lane);
  frag_to_f32(acc, kt[wv], lane);
  // V1 -> vt
  gemm_tile(ah[wv], wfrag + 2 * 16384, v1b, acc, lane);
  frag_to_f32(acc, vt[wv], lane);
  __syncthreads();

  accum_ctx(kt, vt, ctx1 + (size_t)bi * 2048, ksum1 + (size_t)bi * 128, tid);
}

// K2: attn1-apply -> MLP1 -> +resid -> LN2 (x out) -> K2/V2 GEMMs -> ctx2;
//     LN(fx) -> fxn; Q2 GEMM -> softmax -> q2s.
__global__ __launch_bounds__(TPB) void k2_kernel(
    const float* fx, const float* ln2w, const float* ln2b,
    const float* lnfw, const float* lnfb,
    const float* m1b1v, const float* m1b2v,
    const float* q2bv, const float* k2bv, const float* v2bv,
    const _Float16* wfrag, const float* xln1, const float* q1s,
    const float* ctx1, const float* ksum1,
    float* fxn, float* q2s, float* ctx2, float* ksum2, float* xout) {
  __shared__ __align__(16) _Float16 ah[4][2048];
  __shared__ __align__(16) float fs_kt[4][2048];   // q-stage / x2-stage, then k-tile
  __shared__ __align__(16) float vt[4][2048];
  __shared__ __align__(16) float ctxL[2048];
  __shared__ __align__(16) float ksL[128];
  int wv = threadIdx.x >> 5, lane = threadIdx.x & 31, tid = threadIdx.x;
  int tok0 = blockIdx.x * ROWS_PER_WG;
  int tw = tok0 + wv * 16;
  int bi = tok0 / NPERB;

  for (int e = tid; e < 2048; e += TPB) ctxL[e] = ctx1[(size_t)bi * 2048 + e];
  for (int e = tid; e < 128; e += TPB) ksL[e] = ksum1[(size_t)bi * 128 + e];
  stage16(q1s + (size_t)tw * 128, fs_kt[wv], lane);
  __syncthreads();

  attn_apply(fs_kt[wv], ctxL, ksL, ah[wv], lane);
  __syncthreads();

  float acc[8][8];
  // MLP1 layer 1 + GELU
  gemm_tile(ah[wv], wfrag + 6 * 16384, m1b1v, acc, lane);
#pragma unroll
  for (int ct = 0; ct < 8; ++ct)
#pragma unroll
    for (int r = 0; r < 8; ++r) acc[ct][r] = gelu_exact(acc[ct][r]);
  frag_to_f16(acc, ah[wv], lane);
  __syncthreads();
  // MLP1 layer 2 + residual (xln1)
  gemm_tile(ah[wv], wfrag + 7 * 16384, m1b2v, acc, lane);
  frag_add_global(acc, xln1 + (size_t)tw * 128, lane);
  frag_to_f32(acc, fs_kt[wv], lane);
  __syncthreads();
  // LN2 -> x output (f32) + f16 A tile
  ln16(fs_kt[wv], xout + (size_t)tw * 128, ah[wv], ln2w, ln2b, lane);
  __syncthreads();
  // K2 -> softmax -> kt (reuses fs_kt)
  gemm_tile(ah[wv], wfrag + 4 * 16384, k2bv, acc, lane);
#pragma unroll
  for (int ct = 0; ct < 8; ++ct) softmax_ct(acc[ct], lane);
  frag_to_f32(acc, fs_kt[wv], lane);
  // V2 -> vt
  gemm_tile(ah[wv], wfrag + 5 * 16384, v2bv, acc, lane);
  frag_to_f32(acc, vt[wv], lane);
  __syncthreads();
  accum_ctx(fs_kt, vt, ctx2 + (size_t)bi * 2048, ksum2 + (size_t)bi * 128, tid);
  __syncthreads();

  // fx branch: LN(fx) -> fxn (f32) + f16 tile; Q2 GEMM -> softmax -> q2s
  ln16(fx + (size_t)tw * 128, fxn + (size_t)tw * 128, ah[wv], lnfw, lnfb, lane);
  __syncthreads();
  gemm_tile(ah[wv], wfrag + 3 * 16384, q2bv, acc, lane);
#pragma unroll
  for (int ct = 0; ct < 8; ++ct) softmax_ct(acc[ct], lane);
  frag_to_global(acc, q2s + (size_t)tw * 128, lane);
}

// K3: attn2-apply -> MLP2 -> + fxn -> fx_out.
__global__ __launch_bounds__(TPB) void k3_kernel(
    const float* m2b1v, const float* m2b2v, const _Float16* wfrag,
    const float* fxn, const float* q2s, const float* ctx2, const float* ksum2,
    float* fxout) {
  __shared__ __align__(16) _Float16 ah[4][2048];
  __shared__ __align__(16) float fs[4][2048];
  __shared__ __align__(16) float ctxL[2048];
  __shared__ __align__(16) float ksL[128];
  int wv = threadIdx.x >> 5, lane = threadIdx.x & 31, tid = threadIdx.x;
  int tok0 = blockIdx.x * ROWS_PER_WG;
  int tw = tok0 + wv * 16;
  int bi = tok0 / NPERB;

  for (int e = tid; e < 2048; e += TPB) ctxL[e] = ctx2[(size_t)bi * 2048 + e];
  for (int e = tid; e < 128; e += TPB) ksL[e] = ksum2[(size_t)bi * 128 + e];
  stage16(q2s + (size_t)tw * 128, fs[wv], lane);
  __syncthreads();

  attn_apply(fs[wv], ctxL, ksL, ah[wv], lane);
  __syncthreads();

  float acc[8][8];
  gemm_tile(ah[wv], wfrag + 8 * 16384, m2b1v, acc, lane);
#pragma unroll
  for (int ct = 0; ct < 8; ++ct)
#pragma unroll
    for (int r = 0; r < 8; ++r) acc[ct][r] = gelu_exact(acc[ct][r]);
  frag_to_f16(acc, ah[wv], lane);
  __syncthreads();
  gemm_tile(ah[wv], wfrag + 9 * 16384, m2b2v, acc, lane);
  frag_add_global(acc, fxn + (size_t)tw * 128, lane);
  frag_to_global(acc, fxout + (size_t)tw * 128, lane);
}

// ------------------------------ launch -------------------------------------

extern "C" void kernel_launch(void* const* d_in, const int* in_sizes, int n_in,
                              void* d_out, int out_size, void* d_ws, size_t ws_size,
                              hipStream_t stream) {
  (void)in_sizes; (void)n_in; (void)out_size; (void)ws_size;
  const float* x_in = (const float*)d_in[0];
  const float* fx   = (const float*)d_in[1];
  const float* ln1w = (const float*)d_in[2];  const float* ln1b = (const float*)d_in[3];
  const float* ln2w = (const float*)d_in[4];  const float* ln2b = (const float*)d_in[5];
  const float* lnfw = (const float*)d_in[6];  const float* lnfb = (const float*)d_in[7];
  const float* q1w = (const float*)d_in[8];   const float* q1b = (const float*)d_in[9];
  const float* k1w = (const float*)d_in[10];  const float* k1b = (const float*)d_in[11];
  const float* v1w = (const float*)d_in[12];  const float* v1b = (const float*)d_in[13];
  const float* q2w = (const float*)d_in[14];  const float* q2b = (const float*)d_in[15];
  const float* k2w = (const float*)d_in[16];  const float* k2b = (const float*)d_in[17];
  const float* v2w = (const float*)d_in[18];  const float* v2b = (const float*)d_in[19];
  const float* m1w1 = (const float*)d_in[20]; const float* m1b1 = (const float*)d_in[21];
  const float* m1w2 = (const float*)d_in[22]; const float* m1b2 = (const float*)d_in[23];
  const float* m2w1 = (const float*)d_in[24]; const float* m2b1 = (const float*)d_in[25];
  const float* m2w2 = (const float*)d_in[26]; const float* m2b2 = (const float*)d_in[27];

  float* ws = (float*)d_ws;
  _Float16* wfrag = (_Float16*)((char*)d_ws + OFF_WFRAG_BYTES);
  float* xout  = (float*)d_out;
  float* fxout = (float*)d_out + 4194304ull;

  // zero ctx/ksum accumulators (17408 floats)
  zero_kernel<<<(17408 + 255) / 256, 256, 0, stream>>>(ws + OFF_CTX1, 17408);

  WPtrs wp;
  wp.w[0] = q1w; wp.w[1] = k1w; wp.w[2] = v1w; wp.w[3] = q2w; wp.w[4] = k2w;
  wp.w[5] = v2w; wp.w[6] = m1w1; wp.w[7] = m1w2; wp.w[8] = m2w1; wp.w[9] = m2w2;
  wconv_kernel<<<(10 * 16384 + 255) / 256, 256, 0, stream>>>(wp, wfrag);

  k1_kernel<<<NBLK, TPB, 0, stream>>>(
      x_in, ln1w, ln1b, q1b, k1b, v1b, wfrag,
      ws + OFF_XLN1, ws + OFF_Q1S, ws + OFF_CTX1, ws + OFF_KSUM1);

  k2_kernel<<<NBLK, TPB, 0, stream>>>(
      fx, ln2w, ln2b, lnfw, lnfb, m1b1, m1b2, q2b, k2b, v2b, wfrag,
      ws + OFF_XLN1, ws + OFF_Q1S, ws + OFF_CTX1, ws + OFF_KSUM1,
      ws + OFF_FXN, ws + OFF_Q2S, ws + OFF_CTX2, ws + OFF_KSUM2, xout);

  k3_kernel<<<NBLK, TPB, 0, stream>>>(
      m2b1, m2b2, wfrag, ws + OFF_FXN, ws + OFF_Q2S,
      ws + OFF_CTX2, ws + OFF_KSUM2, fxout);
}
